// BDHModel_86242943304235
// MI455X (gfx1250) — matrix operand order repaired and checked by reading, not compiled, not run on hardware
//
#include <hip/hip_runtime.h>
#include <hip/hip_bf16.h>
#include <math.h>

#define HH   3
#define VV   32000
#define DDIM 2048
#define LL   8192
#define TOPK 20

typedef __attribute__((ext_vector_type(16))) _Float16 v16h;
typedef __attribute__((ext_vector_type(8)))  _Float16 v8h;
typedef __attribute__((ext_vector_type(8)))  float    v8f;

// Load 16 contiguous f16 (32 B, 16-byte aligned) as a WMMA fragment.
__device__ __forceinline__ v16h load_frag16(const _Float16* __restrict__ p) {
    v8h lo = *(const v8h*)(p);
    v8h hi = *(const v8h*)(p + 8);
    return __builtin_shufflevector(lo, hi,
        0,1,2,3,4,5,6,7,8,9,10,11,12,13,14,15);
}

// A-fragment: two 8-half chunks 16 halves apart (ISA 16-bit A layout).
__device__ __forceinline__ v16h load_frag_a(const _Float16* __restrict__ p) {
    v8h lo = *(const v8h*)(p);
    v8h hi = *(const v8h*)(p + 16);
    return __builtin_shufflevector(lo, hi,
        0,1,2,3,4,5,6,7,8,9,10,11,12,13,14,15);
}

// ---------------------------------------------------------------------------
// Kernel 1: gather projections[h, tokens[l], :], find 20th-largest threshold
// per row with 20 wave-wide max extractions, write 0/1 activations as f16.
// One wave (32 lanes) per row; 64 values per lane held in registers.
// ---------------------------------------------------------------------------
__global__ __launch_bounds__(256) void topk_acts_kernel(
    const int* __restrict__ tokens,
    const float* __restrict__ proj,
    _Float16* __restrict__ actsH)
{
    const int lane = threadIdx.x & 31;
    const int wave = threadIdx.x >> 5;
    const int rid  = blockIdx.x * 8 + wave;          // [0, H*L)
    if (rid >= HH * LL) return;
    const int h = rid / LL;
    const int l = rid - h * LL;
    const int tok = tokens[l];
    const float* __restrict__ row = proj + ((size_t)h * VV + (size_t)tok) * DDIM;

    float v[64];
#pragma unroll
    for (int i = 0; i < 64; ++i) v[i] = row[lane + 32 * i];

    // Iteratively peel the max 20 times: bound after iteration t == t-th largest.
    float bound = __builtin_inff();
    for (int it = 0; it < TOPK; ++it) {
        float best = -__builtin_inff();
#pragma unroll
        for (int i = 0; i < 64; ++i) {
            float x = v[i];
            best = (x < bound && x > best) ? x : best;
        }
#pragma unroll
        for (int m = 16; m >= 1; m >>= 1) {
            float o = __shfl_xor(best, m, 32);
            best = (o > best) ? o : best;
        }
        bound = best;
    }
    const float thr = bound;

    _Float16* __restrict__ arow = actsH + ((size_t)h * LL + l) * DDIM;
#pragma unroll
    for (int i = 0; i < 64; ++i)
        arow[lane + 32 * i] = (v[i] >= thr) ? (_Float16)1.0f : (_Float16)0.0f;
}

// ---------------------------------------------------------------------------
// Kernel 2: sigmas f32 -> f16, row-major [h][e][d] (== WMMA B-operand order).
// ---------------------------------------------------------------------------
__global__ __launch_bounds__(256) void sigma_cvt_kernel(
    const float* __restrict__ sig, _Float16* __restrict__ sigH, int n)
{
    int i = blockIdx.x * blockDim.x + threadIdx.x;
    const int stride = gridDim.x * blockDim.x;
    for (; i < n; i += stride) sigH[i] = (_Float16)sig[i];
}

// ---------------------------------------------------------------------------
// Kernel 3: fused preds = X * sigma^T via v_wmma_f32_16x16x32_f16, with the
// dot(preds, x_next) and ||preds||^2 reductions fused into the epilogue.
// Block = 128 threads (4 waves) handles 16 rows of one head; wave w owns
// e-range [w*512, w*512+512). 4-way N-tile blocking: each outer step runs
// 4 e-tiles with 4 independent accumulators, so one A-fragment load feeds
// 4 WMMAs (4x A reuse) and the 4 WMMAs form independent dep chains.
// ---------------------------------------------------------------------------
__global__ __launch_bounds__(128) void fused_overlap_kernel(
    const _Float16* __restrict__ actsH,
    const _Float16* __restrict__ sigH,
    float* __restrict__ out)
{
    const int h    = blockIdx.y;
    const int l0   = blockIdx.x * 16;
    const int tid  = threadIdx.x;
    const int lane = tid & 31;
    const int wave = tid >> 5;        // 0..3
    const int nh   = lane & 15;       // lane-in-half: A row / B column index
    const int half = lane >> 4;       // 0 or 1

    const _Float16* __restrict__ X = actsH + (size_t)h * LL * DDIM;
    const _Float16* __restrict__ S = sigH  + (size_t)h * DDIM * DDIM;

    // A row for this lane: row = l0+nh; half selects the K sub-octets.
    const _Float16* __restrict__ Arow = X + (size_t)(l0 + nh) * DDIM + half * 8;

    float dotp[8], n2p[8];
#pragma unroll
    for (int j = 0; j < 8; ++j) { dotp[j] = 0.0f; n2p[j] = 0.0f; }

    const int e_base = wave * 512;

    for (int eo = 0; eo < 8; ++eo) {
        const int e0 = e_base + eo * 64;             // 4 tiles: e0, e0+16, e0+32, e0+48
        // B-fragment rows: lane holds sigma row (e0 + 16*t + nh); half selects the K sub-octets.
        const _Float16* __restrict__ B0 = S + (size_t)(e0 + nh) * DDIM + half * 8;

        v8f c[4];
#pragma unroll
        for (int t = 0; t < 4; ++t) c[t] = (v8f){};

        for (int d0 = 0; d0 < DDIM; d0 += 32) {
            const v16h a = load_frag_a(Arow + d0);
#pragma unroll
            for (int t = 0; t < 4; ++t) {
                const v16h b = load_frag_a(B0 + (size_t)(16 * t) * DDIM + d0);
                c[t] = __builtin_amdgcn_wmma_f32_16x16x32_f16(
                           /*neg_a=*/false, a, /*neg_b=*/false, b,
                           /*c_mod=*/(short)0, c[t],
                           /*reuse_a=*/false, /*reuse_b=*/false);
            }
        }

        // Epilogue: C layout -> VGPR j holds row (l0 + j + 8*half), N = e + nh.
#pragma unroll
        for (int t = 0; t < 4; ++t) {
            const int e = e0 + 16 * t;
#pragma unroll
            for (int j = 0; j < 8; ++j) {
                const int l = l0 + j + 8 * half;
                float xn = 0.0f;
                if (l + 1 < LL)
                    xn = (float)X[(size_t)(l + 1) * DDIM + (e + nh)];
                const float p = c[t][j];
                dotp[j] += p * xn;
                n2p[j]  += p * p;
            }
        }
    }

    // Reduce over the 16 lanes of each half (masks < 16 stay within a half).
#pragma unroll
    for (int j = 0; j < 8; ++j) {
#pragma unroll
        for (int m = 8; m >= 1; m >>= 1) {
            dotp[j] += __shfl_xor(dotp[j], m, 32);
            n2p[j]  += __shfl_xor(n2p[j],  m, 32);
        }
    }

    __shared__ float s_dot[4][16];
    __shared__ float s_n2[4][16];
    if (nh == 0) {
#pragma unroll
        for (int j = 0; j < 8; ++j) {
            s_dot[wave][j + 8 * half] = dotp[j];
            s_n2 [wave][j + 8 * half] = n2p[j];
        }
    }
    __syncthreads();

    if (tid < 16) {
        const int l = l0 + tid;
        if (l < LL - 1) {
            const float dot = s_dot[0][tid] + s_dot[1][tid] + s_dot[2][tid] + s_dot[3][tid];
            const float n2  = s_n2 [0][tid] + s_n2 [1][tid] + s_n2 [2][tid] + s_n2 [3][tid];
            const float denom = sqrtf(n2) * 4.47213595499958f + 1e-8f; // sqrt(K)=sqrt(20)
            out[(size_t)h * (LL - 1) + l] = 1.0f - dot / denom;
        }
    }
}

// ---------------------------------------------------------------------------
extern "C" void kernel_launch(void* const* d_in, const int* in_sizes, int n_in,
                              void* d_out, int out_size, void* d_ws, size_t ws_size,
                              hipStream_t stream)
{
    const int*   tokens = (const int*)d_in[0];
    const float* proj   = (const float*)d_in[1];
    const float* sig    = (const float*)d_in[2];
    float*       out    = (float*)d_out;

    // Workspace layout:
    //   actsH : H*L*D f16  = 100,663,296 B
    //   sigH  : H*D*D f16  =  25,165,824 B
    _Float16* actsH = (_Float16*)d_ws;
    _Float16* sigH  = (_Float16*)((char*)d_ws + (size_t)HH * LL * DDIM * sizeof(_Float16));

    // K1: gather + top-20 threshold + f16 activations. 8 rows per 256-thread block.
    topk_acts_kernel<<<(HH * LL) / 8, 256, 0, stream>>>(tokens, proj, actsH);

    // K2: sigma f32 -> f16.
    sigma_cvt_kernel<<<2048, 256, 0, stream>>>(sig, sigH, HH * DDIM * DDIM);

    // K3: fused WMMA GEMM + overlap output. 512 row-blocks x 3 heads.
    fused_overlap_kernel<<<dim3(512, HH), 128, 0, stream>>>(actsH, sigH, out);
}
